// Net_1331439862307
// MI455X (gfx1250) — matrix operand order repaired and checked by reading, not compiled, not run on hardware
//
#include <hip/hip_runtime.h>
#include <hip/hip_bf16.h>

typedef __attribute__((ext_vector_type(16))) _Float16 v16h;
typedef __attribute__((ext_vector_type(8)))  _Float16 v8h;
typedef __attribute__((ext_vector_type(4)))  _Float16 v4h;
typedef __attribute__((ext_vector_type(8)))  float    v8f;

namespace {
constexpr int   T_STEPS  = 30;
constexpr int   BATCH    = 4096;
constexpr int   D_IN     = 784;
constexpr int   HID      = 128;
constexpr int   OUTD     = 10;
constexpr int   KPAD     = 800;          // 784 padded to 25 * 32
constexpr int   BB       = 64;           // batch rows per workgroup
constexpr int   NTHREADS = 128;          // 4 waves (wave32)
constexpr float BETA     = 0.95f;
constexpr float THR      = 1.0f;
}

union frag16 { v16h v; v8h h[2]; };

__global__ __launch_bounds__(NTHREADS)
void snn_lif_forward(const float* __restrict__ x,
                     const float* __restrict__ W1,
                     const float* __restrict__ b1,
                     const float* __restrict__ W2,
                     const float* __restrict__ b2,
                     float* __restrict__ out_spk,
                     float* __restrict__ out_mem)
{
    extern __shared__ _Float16 smem[];
    _Float16* sW1 = smem;                       // [HID][KPAD]   f16 weights, zero-padded K
    _Float16* sX  = smem + HID * KPAD;          // [BB][KPAD]    x tile; cols 0..127 reused for spk1
    _Float16* sW2 = sX   + BB  * KPAD;          // [16][HID]     rows 10..15 zero

    const int tid  = threadIdx.x;
    const int lane = tid & 31;
    const int wave = tid >> 5;
    const int lo   = lane & 15;                 // A row / B col / C col within tile
    const int kh   = lane >> 4;                 // K-half selector per WMMA layout
    const bool hiL = (kh != 0);
    const int wrow = wave * 16;                 // wave's row offset inside the tile
    const int rowBase = blockIdx.x * BB;

    // ---- one-time: convert W1, W2 to f16 in LDS (B-fragment layout: [n][k]) ----
    for (int i = tid; i < HID * (KPAD / 4); i += NTHREADS) {
        int r = i / (KPAD / 4), q = i % (KPAD / 4);
        v4h o;
        if (q < D_IN / 4) {
            float4 f = reinterpret_cast<const float4*>(W1 + (size_t)r * D_IN)[q];
            o[0] = (_Float16)f.x; o[1] = (_Float16)f.y;
            o[2] = (_Float16)f.z; o[3] = (_Float16)f.w;
        } else {
            o[0] = o[1] = o[2] = o[3] = (_Float16)0.f;
        }
        *reinterpret_cast<v4h*>(&sW1[r * KPAD + q * 4]) = o;
    }
    for (int i = tid; i < 16 * (HID / 4); i += NTHREADS) {
        int r = i / (HID / 4), q = i % (HID / 4);
        v4h o;
        if (r < OUTD) {
            float4 f = reinterpret_cast<const float4*>(W2 + (size_t)r * HID)[q];
            o[0] = (_Float16)f.x; o[1] = (_Float16)f.y;
            o[2] = (_Float16)f.z; o[3] = (_Float16)f.w;
        } else {
            o[0] = o[1] = o[2] = o[3] = (_Float16)0.f;  // pad OUT 10 -> 16
        }
        *reinterpret_cast<v4h*>(&sW2[r * HID + q * 4]) = o;
    }

    // per-lane biases (column-indexed to match C-matrix layout: N = lane&15)
    float b1v[8];
    #pragma unroll
    for (int nt = 0; nt < 8; ++nt) b1v[nt] = b1[nt * 16 + lo];
    const float b2v = (lo < OUTD) ? b2[lo] : 0.f;

    // B-fragment row base addresses in sW1 (one weight column per lane)
    const _Float16* w1base[8];
    #pragma unroll
    for (int nt = 0; nt < 8; ++nt)
        w1base[nt] = &sW1[(nt * 16 + lo) * KPAD + kh * 8];

    // membrane potentials live in WMMA accumulators for the whole kernel
    v8f acc[8] = {};    // layer-1 mem: 16 rows x 128 hidden per wave
    v8f c2     = {};    // layer-2 mem: 16 rows x 16 (10 valid) per wave

    for (int t = 0; t < T_STEPS; ++t) {
        __syncthreads();   // previous timestep finished with sX

        // ---- stage x[t] tile: f32 global -> f16 LDS, K padded to 800 ----
        const float* xt = x + (size_t)t * BATCH * D_IN + (size_t)rowBase * D_IN;
        for (int i = tid; i < BB * (KPAD / 4); i += NTHREADS) {
            int r = i / (KPAD / 4), q = i % (KPAD / 4);
            v4h o;
            if (q < D_IN / 4) {
                float4 f = reinterpret_cast<const float4*>(xt + (size_t)r * D_IN)[q];
                o[0] = (_Float16)f.x; o[1] = (_Float16)f.y;
                o[2] = (_Float16)f.z; o[3] = (_Float16)f.w;
            } else {
                o[0] = o[1] = o[2] = o[3] = (_Float16)0.f;
            }
            *reinterpret_cast<v4h*>(&sX[r * KPAD + q * 4]) = o;
        }
        if (t + 1 < T_STEPS) {  // hint next tile toward L2 (global_prefetch_b8)
            __builtin_prefetch(x + (size_t)(t + 1) * BATCH * D_IN
                                 + (size_t)(rowBase + (tid >> 1)) * D_IN, 0, 1);
        }
        __syncthreads();

        // ---- layer-1 LIF pre-update in accumulators: m = beta*m - reset + b1 ----
        #pragma unroll
        for (int nt = 0; nt < 8; ++nt) {
            #pragma unroll
            for (int e = 0; e < 8; ++e) {
                float m = acc[nt][e];
                float r = (m > THR) ? THR : 0.f;
                acc[nt][e] = BETA * m - r + b1v[nt];
            }
        }

        // ---- cur1: mem1 += x_t @ W1^T via WMMA f16 (25 K-steps x 8 N-tiles) ----
        // Issue all 8 B-fragment loads before the 8 back-to-back WMMAs so the
        // LDS latency of fragments 1..7 hides behind WMMA 0 (partial dscnt waits).
        const int arow = (wrow + lo) * KPAD;
        #pragma unroll 1
        for (int ks = 0; ks < KPAD / 32; ++ks) {
            const int kb = ks * 32;
            frag16 a;   // A: 16x32 f16, per documented VGPR layout
            a.h[0] = *reinterpret_cast<const v8h*>(&sX[arow + kb + kh * 8]);
            a.h[1] = *reinterpret_cast<const v8h*>(&sX[arow + kb + 16 + kh * 8]);
            frag16 b[8];
            #pragma unroll
            for (int nt = 0; nt < 8; ++nt) {
                b[nt].h[0] = *reinterpret_cast<const v8h*>(w1base[nt] + kb);
                b[nt].h[1] = *reinterpret_cast<const v8h*>(w1base[nt] + kb + 16);
            }
            #pragma unroll
            for (int nt = 0; nt < 8; ++nt) {
                acc[nt] = __builtin_amdgcn_wmma_f32_16x16x32_f16(
                    false, a.v, false, b[nt].v, (short)0, acc[nt], false, false);
            }
        }

        // ---- spk1 = (mem1 > THR) as f16, overlaid into this wave's own sX rows ----
        #pragma unroll
        for (int nt = 0; nt < 8; ++nt) {
            const int h = nt * 16 + lo;
            #pragma unroll
            for (int e = 0; e < 8; ++e) {
                const int r = wrow + (hiL ? e + 8 : e);
                sX[r * KPAD + h] = (acc[nt][e] > THR) ? (_Float16)1.f : (_Float16)0.f;
            }
        }

        // ---- layer-2 LIF pre-update ----
        #pragma unroll
        for (int e = 0; e < 8; ++e) {
            float m = c2[e];
            float r = (m > THR) ? THR : 0.f;
            c2[e] = BETA * m - r + b2v;
        }

        // ---- cur2: mem2 += spk1 @ W2^T (4 K-steps, N padded to 16) ----
        {
            frag16 a[4], b[4];
            #pragma unroll
            for (int ks = 0; ks < HID / 32; ++ks) {
                const int kb = ks * 32;
                a[ks].h[0] = *reinterpret_cast<const v8h*>(&sX[arow + kb + kh * 8]);
                a[ks].h[1] = *reinterpret_cast<const v8h*>(&sX[arow + kb + 16 + kh * 8]);
                const int woff = lo * HID + kb + kh * 8;
                b[ks].h[0] = *reinterpret_cast<const v8h*>(&sW2[woff]);
                b[ks].h[1] = *reinterpret_cast<const v8h*>(&sW2[woff + 16]);
            }
            #pragma unroll
            for (int ks = 0; ks < HID / 32; ++ks) {
                c2 = __builtin_amdgcn_wmma_f32_16x16x32_f16(
                    false, a[ks].v, false, b[ks].v, (short)0, c2, false, false);
            }
        }

        // ---- emit spk2 / mem2 records (C layout: col = lane&15, row = vgpr) ----
        if (lo < OUTD) {
            #pragma unroll
            for (int e = 0; e < 8; ++e) {
                const int r = rowBase + wrow + (hiL ? e + 8 : e);
                const size_t o = ((size_t)t * BATCH + r) * OUTD + lo;
                out_spk[o] = (c2[e] > THR) ? 1.f : 0.f;
                out_mem[o] = c2[e];
            }
        }
    }
}

extern "C" void kernel_launch(void* const* d_in, const int* in_sizes, int n_in,
                              void* d_out, int out_size, void* d_ws, size_t ws_size,
                              hipStream_t stream) {
    (void)in_sizes; (void)n_in; (void)out_size; (void)d_ws; (void)ws_size;
    const float* x  = (const float*)d_in[0];
    const float* W1 = (const float*)d_in[1];
    const float* b1 = (const float*)d_in[2];
    const float* W2 = (const float*)d_in[3];
    const float* b2 = (const float*)d_in[4];
    float* out_spk = (float*)d_out;
    float* out_mem = out_spk + (size_t)T_STEPS * BATCH * OUTD;

    const size_t shmem =
        (size_t)(HID * KPAD + BB * KPAD + 16 * HID) * sizeof(_Float16);  // 311,296 B
    (void)hipFuncSetAttribute(reinterpret_cast<const void*>(snn_lif_forward),
                              hipFuncAttributeMaxDynamicSharedMemorySize,
                              (int)shmem);
    snn_lif_forward<<<dim3(BATCH / BB), dim3(NTHREADS), shmem, stream>>>(
        x, W1, b1, W2, b2, out_spk, out_mem);
}
